// EnhancedEdgeLayer_21655225106457
// MI455X (gfx1250) — matrix operand order repaired and checked by reading, not compiled, not run on hardware
//
#include <hip/hip_runtime.h>
#include <stdint.h>

// CDNA5 / gfx1250 fused EnhancedEdgeLayer, round 3:
//  - prep kernel packs w1/w2 into WMMA-B-layout f16 matrices in d_ws
//  - async global->LDS (GLOBAL_LOAD_ASYNC_TO_LDS_B128) for the x tile
//  - stencil bank (VALU) -> f16 features, channel-stride 16
//  - conv1 9->32: WMMA f16, K = tap*16+c (5 K-steps), A via ds_load_b128, B via v8h global loads
//  - conv2 32->16: WMMA f16, K = tap*32+c (9 K-steps), A via ds_load_b128
//  - + 1x1 residual, vectorized LDS/global stores

typedef __attribute__((ext_vector_type(16))) _Float16 v16h;
typedef __attribute__((ext_vector_type(8)))  _Float16 v8h;
typedef __attribute__((ext_vector_type(8)))  float    v8f;
typedef __attribute__((ext_vector_type(4)))  float    v4f;

union AFrag { v16h v; v8h h[2]; _Float16 e[16]; };
union CFrag { v8f  v; float e[8]; };

#define IMG_H 512
#define IMG_W 512
#define NB    8
#define SXS   28            // x-tile row stride (floats): 16B-aligned quads
#define W1P_ELEMS (32 * 160)   // [n][K], K = tap*16 + c (10 taps x 16 ch)
#define W2P_ELEMS (16 * 288)   // [n][K], K = tap*32 + c (9 taps x 32 ch)

// ---- Prep: pack weights into WMMA-B fragment layout (f16) ------------------
__global__ __launch_bounds__(256)
void pack_weights_kernel(const float* __restrict__ w1, const float* __restrict__ w2,
                         _Float16* __restrict__ w1p, _Float16* __restrict__ w2p)
{
    const int tid = threadIdx.x;
    for (int i = tid; i < W1P_ELEMS; i += 256) {
        int n = i / 160, K = i % 160;
        int tap = K >> 4, c = K & 15;
        float v = (tap < 9 && c < 9) ? w1[n * 81 + c * 9 + tap] : 0.0f;
        w1p[i] = (_Float16)v;
    }
    for (int i = tid; i < W2P_ELEMS; i += 256) {
        int n = i / 288, K = i % 288;
        int tp = K >> 5, c = K & 31;
        w2p[i] = (_Float16)w2[n * 288 + c * 9 + tp];
    }
}

// ---- Main fused kernel -----------------------------------------------------
__global__ __launch_bounds__(256)
void edge_fused_kernel(const float* __restrict__ x,
                       const _Float16* __restrict__ w1p, const float* __restrict__ b1,
                       const _Float16* __restrict__ w2p, const float* __restrict__ b2,
                       const float* __restrict__ wr, const float* __restrict__ br,
                       float* __restrict__ out)
{
    __shared__ float    sx[24 * SXS];         // x tile + halo 4
    __shared__ _Float16 sf[20 * 20 * 16];     // 9-ch features (+7 zero pad), stride 16
    __shared__ _Float16 sh[336 * 32];         // 18x18 hidden + 12 pad pixel slots
    __shared__ float    sout[16 * 16 * 16];   // [n][py][px] output staging

    const int tid  = threadIdx.x;
    const int lane = tid & 31;
    const int wave = tid >> 5;
    const int hh   = lane >> 4;               // wave32 lane-half for WMMA striping
    const int ln   = lane & 15;

    const int blk = blockIdx.x;
    const int b   = blk >> 10;
    const int tt  = blk & 1023;
    const int ty  = tt >> 5, tx = tt & 31;
    const int oy  = ty * 16, ox = tx * 16;

    const float* xb = x + (size_t)b * (IMG_H * IMG_W);

    if (tid < 16) {
        __builtin_prefetch(&w1p[tid * 320], 0, 0);
        __builtin_prefetch(&w2p[tid * 288], 0, 0);
    }

    // ---- Stage 1: 24x24 x tile (halo 4) ------------------------------------
    const bool interior = (ty >= 1) && (ty <= 30) && (tx >= 1) && (tx <= 30);
    if (interior) {
        if (tid < 144) {          // 144 lanes x 16B quads, ASYNCcnt-tracked
            int r  = tid / 6;
            int c0 = (tid % 6) * 4;
            const float* gp = xb + (size_t)(oy - 4 + r) * IMG_W + (ox - 4 + c0);
            unsigned           loff = (unsigned)(uintptr_t)(&sx[r * SXS + c0]);
            unsigned long long ga   = (unsigned long long)(uintptr_t)gp;
            asm volatile("global_load_async_to_lds_b128 %0, %1, off"
                         :: "v"(loff), "v"(ga) : "memory");
        }
        asm volatile("s_wait_asynccnt 0x0" ::: "memory");
    } else {
        for (int i = tid; i < 24 * 24; i += 256) {
            int r = i / 24, c = i % 24;
            int gy = oy - 4 + r, gx = ox - 4 + c;
            float v = 0.0f;
            if (gy >= 0 && gy < IMG_H && gx >= 0 && gx < IMG_W) v = xb[gy * IMG_W + gx];
            sx[r * SXS + c] = v;
        }
    }
    __syncthreads();

    // ---- Stage 2: 8 stencils + x -> 16-ch-padded f16 features (20x20) ------
    for (int i = tid; i < 400; i += 256) {
        int iy = i / 20, ix = i % 20;
        float t[5][5];
        #pragma unroll
        for (int u = 0; u < 5; ++u)
            #pragma unroll
            for (int v = 0; v < 5; ++v)
                t[u][v] = sx[(iy + u) * SXS + (ix + v)];

        float sobx = -t[1][1] + t[1][3] - 2.f*t[2][1] + 2.f*t[2][3] - t[3][1] + t[3][3];
        float soby = -t[1][1] - 2.f*t[1][2] - t[1][3] + t[3][1] + 2.f*t[3][2] + t[3][3];
        float dg1  =  t[1][2] + 2.f*t[1][3] - t[2][1] + t[2][3] - 2.f*t[3][1] - t[3][2];
        float dg2  = -2.f*t[1][1] - t[1][2] - t[2][1] + t[2][3] + t[3][2] + 2.f*t[3][3];
        float lap  =  t[1][2] + t[2][1] - 4.f*t[2][2] + t[2][3] + t[3][2];
        float g5   = -1.f*t[0][0] - 2.f*t[0][1] + 2.f*t[0][3] + 1.f*t[0][4]
                   + -2.f*t[1][0] - 3.f*t[1][1] + 3.f*t[1][3] + 2.f*t[1][4]
                   + -3.f*t[2][0] - 4.f*t[2][1] + 4.f*t[2][3] + 3.f*t[2][4]
                   + -2.f*t[3][0] - 3.f*t[3][1] + 3.f*t[3][3] + 2.f*t[3][4]
                   + -1.f*t[4][0] - 2.f*t[4][1] + 2.f*t[4][3] + 1.f*t[4][4];
        float scx  = -3.f*t[1][1] + 3.f*t[1][3] - 10.f*t[2][1] + 10.f*t[2][3] - 3.f*t[3][1] + 3.f*t[3][3];
        float scy  = -3.f*t[1][1] - 10.f*t[1][2] - 3.f*t[1][3] + 3.f*t[3][1] + 10.f*t[3][2] + 3.f*t[3][3];

        union { v8h v[2]; _Float16 e[16]; } fv;
        fv.e[0] = (_Float16)sobx; fv.e[1] = (_Float16)soby;
        fv.e[2] = (_Float16)dg1;  fv.e[3] = (_Float16)dg2;
        fv.e[4] = (_Float16)lap;  fv.e[5] = (_Float16)g5;
        fv.e[6] = (_Float16)scx;  fv.e[7] = (_Float16)scy;
        fv.e[8] = (_Float16)t[2][2];
        #pragma unroll
        for (int c = 9; c < 16; ++c) fv.e[c] = (_Float16)0.0f;

        _Float16* fp = &sf[(iy * 20 + ix) * 16];
        *(v8h*)(fp)     = fv.v[0];
        *(v8h*)(fp + 8) = fv.v[1];
    }
    __syncthreads();

    // ---- Stage 3: conv1 (9->32, 3x3), K = tap*16 + c, 5 K-steps ------------
    AFrag B1[2][5];
    #pragma unroll
    for (int nf = 0; nf < 2; ++nf)
        #pragma unroll
        for (int s = 0; s < 5; ++s)
        {
            B1[nf][s].h[0] = *(const v8h*)(w1p + (nf * 16 + ln) * 160 + s * 32 + hh * 8);
            B1[nf][s].h[1] = *(const v8h*)(w1p + (nf * 16 + ln) * 160 + s * 32 + 16 + hh * 8);
        }

    const float bias0 = b1[ln];
    const float bias1 = b1[16 + ln];
    const v8h zero8 = {};

    for (int f = wave; f < 21; f += 8) {
        int pA = f * 16 + ln;
        if (pA > 323) pA = 323;               // clamp A rows of the ragged tail
        int gy = pA / 18, gx = pA % 18;

        CFrag C0, C1;
        #pragma unroll
        for (int r = 0; r < 8; ++r) { C0.e[r] = 0.0f; C1.e[r] = 0.0f; }

        #pragma unroll
        for (int s = 0; s < 5; ++s) {
            const int tap0 = 2 * s, tap1 = 2 * s + 1;
            const int dy0 = tap0 / 3, dx0 = tap0 % 3;
            const int dy1 = tap1 / 3, dx1 = tap1 % 3;
            AFrag A;
            A.h[0] = *(const v8h*)&sf[((gy + dy0) * 20 + (gx + dx0)) * 16 + hh * 8];
            if (tap1 < 9)
                A.h[1] = *(const v8h*)&sf[((gy + dy1) * 20 + (gx + dx1)) * 16 + hh * 8];
            else
                A.h[1] = zero8;

            C0.v = __builtin_amdgcn_wmma_f32_16x16x32_f16(false, A.v, false, B1[0][s].v,
                                                          (short)0, C0.v, false, false);
            C1.v = __builtin_amdgcn_wmma_f32_16x16x32_f16(false, A.v, false, B1[1][s].v,
                                                          (short)0, C1.v, false, false);
        }

        // unguarded epilogue: ragged rows land in sh padding (pixels 324..335)
        #pragma unroll
        for (int r = 0; r < 8; ++r) {
            int p = f * 16 + hh * 8 + r;      // sh[(p/18)*18 + p%18] == sh[p]
            float v0 = C0.e[r] + bias0; v0 = v0 > 0.0f ? v0 : 0.0f;
            float v1 = C1.e[r] + bias1; v1 = v1 > 0.0f ? v1 : 0.0f;
            sh[p * 32 + ln]      = (_Float16)v0;
            sh[p * 32 + 16 + ln] = (_Float16)v1;
        }
    }
    __syncthreads();

    // ---- Stage 4: conv2 (32->16, 3x3), K = tap*32 + c, 9 K-steps -----------
    AFrag B2[9];
    #pragma unroll
    for (int tp = 0; tp < 9; ++tp)
    {
        B2[tp].h[0] = *(const v8h*)(w2p + ln * 288 + tp * 32 + hh * 8);
        B2[tp].h[1] = *(const v8h*)(w2p + ln * 288 + tp * 32 + 16 + hh * 8);
    }

    const float bias2 = b2[ln];
    const float wrv = wr[ln];
    const float brv = br[ln];

    for (int f = wave; f < 16; f += 8) {
        CFrag C;
        #pragma unroll
        for (int r = 0; r < 8; ++r) C.e[r] = 0.0f;

        #pragma unroll
        for (int tp = 0; tp < 9; ++tp) {
            const int dy = tp / 3, dx = tp % 3;
            const int pix = (f + dy) * 18 + (ln + dx);
            AFrag A;
            A.h[0] = *(const v8h*)&sh[pix * 32 + hh * 8];
            A.h[1] = *(const v8h*)&sh[pix * 32 + 16 + hh * 8];
            C.v = __builtin_amdgcn_wmma_f32_16x16x32_f16(false, A.v, false, B2[tp].v,
                                                         (short)0, C.v, false, false);
        }

        union { v4f q[2]; float e[8]; } res;
        #pragma unroll
        for (int r = 0; r < 8; ++r) {
            int qx = hh * 8 + r;
            float xv = sx[(f + 4) * SXS + (qx + 4)];
            float v = C.e[r] + bias2;
            v = v > 0.0f ? v : 0.0f;
            res.e[r] = v + wrv * xv + brv;
        }
        float* op = &sout[ln * 256 + f * 16 + hh * 8];
        *(v4f*)(op)     = res.q[0];           // ds_store_b128
        *(v4f*)(op + 4) = res.q[1];           // ds_store_b128
    }
    __syncthreads();

    // ---- Stage 5: vectorized coalesced store -------------------------------
    for (int i = tid; i < 1024; i += 256) {   // 1024 float4 quads
        int n   = i >> 6;
        int rem = i & 63;
        int py  = rem >> 2, q = rem & 3;
        v4f val = *(const v4f*)&sout[n * 256 + py * 16 + q * 4];
        *(v4f*)&out[(((size_t)b * 16 + n) * IMG_H + (oy + py)) * IMG_W + ox + q * 4] = val;
    }
}

extern "C" void kernel_launch(void* const* d_in, const int* in_sizes, int n_in,
                              void* d_out, int out_size, void* d_ws, size_t ws_size,
                              hipStream_t stream) {
    (void)in_sizes; (void)n_in; (void)ws_size; (void)out_size;
    const float* x  = (const float*)d_in[0];
    const float* w1 = (const float*)d_in[1];
    const float* b1 = (const float*)d_in[2];
    const float* w2 = (const float*)d_in[3];
    const float* b2 = (const float*)d_in[4];
    const float* wr = (const float*)d_in[5];
    const float* br = (const float*)d_in[6];
    float* out = (float*)d_out;

    _Float16* w1p = (_Float16*)d_ws;             // 5120 halfs
    _Float16* w2p = w1p + W1P_ELEMS;             // 4608 halfs (total ~19.5 KB of d_ws)

    hipLaunchKernelGGL(pack_weights_kernel, dim3(1), dim3(256), 0, stream, w1, w2, w1p, w2p);

    dim3 grid(NB * 32 * 32);   // 8192 workgroups, one per 16x16 output tile
    dim3 block(256);           // 8 wave32 waves
    hipLaunchKernelGGL(edge_fused_kernel, grid, block, 0, stream,
                       x, w1p, b1, w2p, b2, wr, br, out);
}
